// GlobalAttention_48533130445227
// MI455X (gfx1250) — matrix operand order repaired and checked by reading, not compiled, not run on hardware
//
#include <hip/hip_runtime.h>
#include <hip/hip_bf16.h>
#include <math.h>

// Problem constants (match the reference)
#define N_TOK  8192
#define DMODEL 1024
#define D3     3072
#define NH     16
#define NG     16
#define SEQ    512   // tokens per group
#define HDIM   64    // head dim

typedef __attribute__((ext_vector_type(16))) __bf16 v16bf;
typedef __attribute__((ext_vector_type(8)))  __bf16 v8bf;
typedef __attribute__((ext_vector_type(8)))  float  v8f;

__device__ __forceinline__ int lane_id() { return (int)(threadIdx.x & 31u); }

__device__ __forceinline__ v16bf cat8(v8bf lo, v8bf hi) {
  return __builtin_shufflevector(lo, hi, 0,1,2,3,4,5,6,7,8,9,10,11,12,13,14,15);
}

// ---- WMMA fragment loaders (wave32, v_wmma_f32_16x16x32_bf16) ----
// A matrix 16x32 (MxK), row-major source, base at (row0,k0), stride lda.
// Lane layout (ISA 7.12.2): lane = 16*kh + m; element i -> K = i + 8*kh + (i>=8 ? 8 : 0)
// => per lane: two contiguous 8xbf16 (16B) loads at k-offsets 8*kh and 16+8*kh.
__device__ __forceinline__ v16bf load_a_bf16(const __bf16* __restrict__ base, int lda) {
  const int m = lane_id() & 15, kh = lane_id() >> 4;
  const __bf16* p = base + (size_t)m * lda + 8 * kh;
  v8bf lo = *(const v8bf*)(p);
  v8bf hi = *(const v8bf*)(p + 16);
  return cat8(lo, hi);
}

// B matrix 32x16 (KxN) where logical B[k][n] = W[n][k], W row-major stride ldw
// (i.e. computing X * W^T). Lane layout: lane = 16*kh + n; element i -> K = i + 8*kh + (i>=8 ? 8 : 0)
// => two contiguous 8xbf16 (16B) loads per lane at k-offsets 8*kh and 16+8*kh.
__device__ __forceinline__ v16bf load_bT_bf16(const __bf16* __restrict__ base, int ldw) {
  const int n = lane_id() & 15, kh = lane_id() >> 4;
  const __bf16* p = base + (size_t)n * ldw + 8 * kh;
  v8bf lo = *(const v8bf*)(p);
  v8bf hi = *(const v8bf*)(p + 16);
  return cat8(lo, hi);
}

// A fragment gathered from fp32 LDS (softmax probabilities), converted to bf16.
__device__ __forceinline__ v16bf load_a_f32(const float* __restrict__ base, int lda) {
  const int m = lane_id() & 15, kh = lane_id() >> 4;
  const float* p = base + (size_t)m * lda + 8 * kh;
  v16bf a;
  #pragma unroll
  for (int i = 0; i < 8; ++i) { a[i] = (__bf16)p[i]; a[i + 8] = (__bf16)p[i + 16]; }
  return a;
}

__device__ __forceinline__ v8f wmma_bf16(v16bf a, v16bf b, v8f c) {
  return __builtin_amdgcn_wmma_f32_16x16x32_bf16(false, a, false, b, (short)0, c,
                                                 false, false);
}

// ---------------- Kernel 1: fp32 -> bf16 conversion ----------------
__global__ void k_cvt_bf16(const float* __restrict__ in, __bf16* __restrict__ out, int n) {
  int i = blockIdx.x * blockDim.x + threadIdx.x;
  const int stride = gridDim.x * blockDim.x;
  for (; i < n; i += stride) out[i] = (__bf16)in[i];
}

// ---------------- Kernel 2: QKV projection GEMM + bias ----------------
// Block = 16 tokens x 1024 columns; 8 waves x 8 tiles. Per K-step: 1 A-fragment
// load + 8 batched B-fragment loads (distinct regs, one clause) -> 8 WMMAs can
// overlap the tail of the load clause instead of draining loadcnt each time.
// Also writes transposed V copy vT[g][h][hd][s] for contiguous attn@V operands.
__global__ __launch_bounds__(256)
void k_qkv_gemm(const __bf16* __restrict__ xb, const __bf16* __restrict__ wb,
                const float* __restrict__ bias,
                __bf16* __restrict__ qkv, __bf16* __restrict__ vT) {
  const int mbase  = blockIdx.x * 16;                      // token tile
  const int wave   = threadIdx.x >> 5;
  const int colblk = blockIdx.y * 1024 + wave * 128;       // 8 tiles of 16 cols
  const int n = lane_id() & 15, kh = lane_id() >> 4;

  v8f acc[8] = {};
  const __bf16* abase = xb + (size_t)mbase * DMODEL;
  const __bf16* bbase = wb + (size_t)colblk * DMODEL;
  for (int kb = 0; kb < DMODEL; kb += 32) {
    v16bf a = load_a_bf16(abase + kb, DMODEL);
    v16bf b[8];
    #pragma unroll
    for (int t = 0; t < 8; ++t)
      b[t] = load_bT_bf16(bbase + (size_t)t * 16 * DMODEL + kb, DMODEL);
    if (kb + 32 < DMODEL)  // pull next A slab toward the WGP (global_prefetch_b8)
      __builtin_prefetch(abase + (size_t)(lane_id() & 15) * DMODEL + kb + 32, 0, 3);
    #pragma unroll
    for (int t = 0; t < 8; ++t)
      acc[t] = wmma_bf16(a, b[t], acc[t]);
  }

  #pragma unroll
  for (int t = 0; t < 8; ++t) {
    const int col = colblk + t * 16 + n;
    const float bval = bias[col];
    #pragma unroll
    for (int r = 0; r < 8; ++r) {
      const int row = mbase + r + 8 * kh;
      const float y = acc[t][r] + bval;
      qkv[(size_t)row * D3 + col] = (__bf16)y;
      if (col >= 2 * DMODEL) {                 // V third -> transposed copy
        const int rel = col - 2 * DMODEL;
        const int h = rel >> 6, hd = rel & 63;
        const int g = row >> 9, s = row & 511; // SEQ = 512
        vT[(((size_t)(g * NH + h) * HDIM) + hd) * SEQ + s] = (__bf16)y;
      }
    }
  }
}

// ---------------- Kernel 3: attention for one (g, h, 16-query tile) ----------------
__global__ __launch_bounds__(128)
void k_attention(const __bf16* __restrict__ qkv, const __bf16* __restrict__ vT,
                 __bf16* __restrict__ ctx) {
  __shared__ float probs[16 * SEQ];   // 32 KB score/prob stripe
  __shared__ float redmax[16][8];
  __shared__ float redsum[16][8];

  const int qt = blockIdx.x & 31;          // query tile in group (SEQ/16 = 32)
  const int h  = (blockIdx.x >> 5) & 15;
  const int g  = blockIdx.x >> 9;
  const int qtok = g * SEQ + qt * 16;
  const int wave = threadIdx.x >> 5;       // 4 waves
  const int n = lane_id() & 15, kh = lane_id() >> 4;

  // Phase 1: scores[16][512] = (q . k^T) * 1/sqrt(HD)
  // q fragments are invariant across key tiles -> hoist them.
  const __bf16* qbase = qkv + (size_t)qtok * D3 + h * HDIM;
  const v16bf aq0 = load_a_bf16(qbase + 0, D3);
  const v16bf aq1 = load_a_bf16(qbase + 32, D3);
  for (int kt = wave; kt < SEQ / 16; kt += 4) {
    const int ktok = g * SEQ + kt * 16;
    const __bf16* kbase = qkv + (size_t)ktok * D3 + DMODEL + h * HDIM; // k block
    v16bf b0 = load_bT_bf16(kbase + 0, D3);
    v16bf b1 = load_bT_bf16(kbase + 32, D3);
    v8f c = {};
    c = wmma_bf16(aq0, b0, c);
    c = wmma_bf16(aq1, b1, c);
    #pragma unroll
    for (int r = 0; r < 8; ++r)
      probs[(r + 8 * kh) * SEQ + kt * 16 + n] = c[r] * 0.125f; // 1/sqrt(64)
  }
  __syncthreads();

  // Phase 2: softmax across each row of 512 (8 threads per row, 64 cols each)
  const int row  = threadIdx.x >> 3;
  const int part = threadIdx.x & 7;
  float* prow = probs + row * SEQ + part * 64;
  float mx = -3.4e38f;
  for (int i = 0; i < 64; ++i) mx = fmaxf(mx, prow[i]);
  redmax[row][part] = mx;
  __syncthreads();
  mx = redmax[row][0];
  #pragma unroll
  for (int i = 1; i < 8; ++i) mx = fmaxf(mx, redmax[row][i]);
  float sum = 0.f;
  for (int i = 0; i < 64; ++i) { float e = __expf(prow[i] - mx); prow[i] = e; sum += e; }
  redsum[row][part] = sum;
  __syncthreads();
  sum = 0.f;
  #pragma unroll
  for (int i = 0; i < 8; ++i) sum += redsum[row][i];
  const float inv = 1.0f / sum;
  for (int i = 0; i < 64; ++i) prow[i] *= inv;
  __syncthreads();

  // Phase 3: ctx[16][64] = probs @ V ; wave w owns hd cols [16w, 16w+16)
  const __bf16* vgh = vT + (size_t)(g * NH + h) * HDIM * SEQ; // [hd][s] row-major
  v8f c = {};
  #pragma unroll 4
  for (int kb = 0; kb < SEQ; kb += 32) {
    v16bf a = load_a_f32(probs + kb, SEQ);
    v16bf b = load_bT_bf16(vgh + (size_t)(wave * 16) * SEQ + kb, SEQ); // B[s][hd]=vT[hd][s]
    c = wmma_bf16(a, b, c);
  }
  #pragma unroll
  for (int r = 0; r < 8; ++r) {
    const int trow = qtok + r + 8 * kh;
    ctx[(size_t)trow * DMODEL + h * HDIM + wave * 16 + n] = (__bf16)c[r];
  }
}

// ---------------- Kernel 4: out-proj GEMM + residual + LayerNorm ----------------
// One block per 16 tokens; 8 waves x 8 tiles = full 1024 output columns.
__global__ __launch_bounds__(256)
void k_out_ln(const __bf16* __restrict__ ctxb, const __bf16* __restrict__ wb,
              const float* __restrict__ x, const float* __restrict__ ob,
              const float* __restrict__ lng, const float* __restrict__ lnb,
              float* __restrict__ out) {
  __shared__ float rowsum[16];
  __shared__ float rowsq[16];
  const int mbase = blockIdx.x * 16;
  const int wave  = threadIdx.x >> 5;
  const int n = lane_id() & 15, kh = lane_id() >> 4;

  if (threadIdx.x < 16) { rowsum[threadIdx.x] = 0.f; rowsq[threadIdx.x] = 0.f; }
  __syncthreads();

  v8f acc[8] = {};
  const __bf16* abase = ctxb + (size_t)mbase * DMODEL;
  const __bf16* bbase = wb + (size_t)(wave * 128) * DMODEL;
  for (int kb = 0; kb < DMODEL; kb += 32) {
    v16bf a = load_a_bf16(abase + kb, DMODEL);
    v16bf b[8];
    #pragma unroll
    for (int t = 0; t < 8; ++t)
      b[t] = load_bT_bf16(bbase + (size_t)t * 16 * DMODEL + kb, DMODEL);
    if (kb + 32 < DMODEL)
      __builtin_prefetch(abase + (size_t)(lane_id() & 15) * DMODEL + kb + 32, 0, 3);
    #pragma unroll
    for (int t = 0; t < 8; ++t)
      acc[t] = wmma_bf16(a, b[t], acc[t]);
  }

  // bias + residual; accumulate LN partial stats in registers
  float psum[8] = {0,0,0,0,0,0,0,0};
  float psq [8] = {0,0,0,0,0,0,0,0};
  #pragma unroll
  for (int t = 0; t < 8; ++t) {
    const int col = wave * 128 + t * 16 + n;
    const float bc = ob[col];
    #pragma unroll
    for (int r = 0; r < 8; ++r) {
      const int row = mbase + r + 8 * kh;
      const float y = acc[t][r] + bc + x[(size_t)row * DMODEL + col];
      acc[t][r] = y;
      psum[r] += y;
      psq [r] += y * y;
    }
  }
  #pragma unroll
  for (int r = 0; r < 8; ++r) {
    atomicAdd(&rowsum[r + 8 * kh], psum[r]);   // ds_add_f32
    atomicAdd(&rowsq [r + 8 * kh], psq [r]);
  }
  __syncthreads();

  float mu[8], rstd[8];
  #pragma unroll
  for (int r = 0; r < 8; ++r) {
    const float m = rowsum[r + 8 * kh] * (1.0f / DMODEL);
    const float v = rowsq [r + 8 * kh] * (1.0f / DMODEL) - m * m;
    mu[r] = m;
    rstd[r] = rsqrtf(v + 1e-5f);
  }
  #pragma unroll
  for (int t = 0; t < 8; ++t) {
    const int col = wave * 128 + t * 16 + n;
    const float gcol = lng[col], bcol = lnb[col];
    #pragma unroll
    for (int r = 0; r < 8; ++r) {
      const int row = mbase + r + 8 * kh;
      out[(size_t)row * DMODEL + col] = (acc[t][r] - mu[r]) * rstd[r] * gcol + bcol;
    }
  }
}

// ---------------- Host-side launcher ----------------
extern "C" void kernel_launch(void* const* d_in, const int* in_sizes, int n_in,
                              void* d_out, int out_size, void* d_ws, size_t ws_size,
                              hipStream_t stream) {
  (void)in_sizes; (void)n_in; (void)out_size; (void)ws_size;
  const float* x     = (const float*)d_in[0];
  // d_in[1] = batch (int64) -- implied by group structure, unused
  const float* w_in  = (const float*)d_in[2];
  const float* b_in  = (const float*)d_in[3];
  const float* w_out = (const float*)d_in[4];
  const float* b_out = (const float*)d_in[5];
  const float* lng   = (const float*)d_in[6];
  const float* lnb   = (const float*)d_in[7];
  float* out = (float*)d_out;

  // Workspace layout (88 MB total):
  char* ws = (char*)d_ws;
  __bf16* xb  = (__bf16*)(ws);                           // 16 MB (reused as ctx)
  __bf16* wib = (__bf16*)(ws + (size_t)(16u << 20));     //  6 MB
  __bf16* wob = (__bf16*)(ws + (size_t)(22u << 20));     //  2 MB
  __bf16* qkv = (__bf16*)(ws + (size_t)(24u << 20));     // 48 MB
  __bf16* vT  = (__bf16*)(ws + (size_t)(72u << 20));     // 16 MB
  __bf16* ctx = xb;  // x_bf dead after kernel 2; reuse for ctx

  k_cvt_bf16<<<1024, 256, 0, stream>>>(x,     xb,  N_TOK * DMODEL);
  k_cvt_bf16<<<1024, 256, 0, stream>>>(w_in,  wib, D3 * DMODEL);
  k_cvt_bf16<<<512,  256, 0, stream>>>(w_out, wob, DMODEL * DMODEL);

  dim3 g2(N_TOK / 16, 3);                 // 16-token rows x 1024-col slabs
  k_qkv_gemm<<<g2, 256, 0, stream>>>(xb, wib, b_in, qkv, vT);

  k_attention<<<NG * NH * (SEQ / 16), 128, 0, stream>>>(qkv, vT, ctx);

  k_out_ln<<<N_TOK / 16, 256, 0, stream>>>(ctx, wob, x, b_out, lng, lnb, out);
}
